// MultiHeadAttention_86268713108156
// MI455X (gfx1250) — matrix operand order repaired and checked by reading, not compiled, not run on hardware
//
#include <hip/hip_runtime.h>
#include <cstdint>

typedef __bf16 bf16;
typedef __attribute__((ext_vector_type(16))) __bf16 v16bf;
typedef __attribute__((ext_vector_type(8)))  __bf16 v8bf;
typedef __attribute__((ext_vector_type(4)))  __bf16 v4bf;
typedef __attribute__((ext_vector_type(8)))  float  v8f;

#define H_SIZE 1024
#define NHEADS 16
#define DK     64
#define LSEQ   2048
#define BATCH  2
#define BL     (BATCH * LSEQ)   // 4096
#define WPITCH 40               // LDS tile row pitch (elements): 32 + 8 pad, 80B (16B aligned)

// ---------------------------------------------------------------------------
// WMMA fragment helpers (layouts per CDNA5 ISA 7.12.2, wave32)
// A (16x32 bf16): lane l holds row (l&15); lanes 0-15 K in {0..7,16..23},
//                 lanes 16-31 K in {8..15,24..31} -> two 16B chunks at +0,+16.
// B (32x16 bf16): lane l holds column (l&15); lanes 0-15 K in {0..7,16..23},
//                 lanes 16-31 K in {8..15,24..31} -> two 16B chunks at +0,+16 (src row-major [N][K]).
// C/D (16x16 f32): VGPR r, lane l -> row r + 8*(l>>4), col (l&15).
// ---------------------------------------------------------------------------
__device__ inline v16bf load_a_frag(const bf16* base, int ld) {
  const int l = threadIdx.x & 31;
  const bf16* p = base + (size_t)(l & 15) * ld + ((l >> 4) << 3);
  v8bf lo = *(const v8bf*)p;
  v8bf hi = *(const v8bf*)(p + 16);
  v16bf r;
#pragma unroll
  for (int i = 0; i < 8; ++i) { r[i] = lo[i]; r[i + 8] = hi[i]; }
  return r;
}

__device__ inline v16bf load_b_frag(const bf16* base, int ld) {
  const int l = threadIdx.x & 31;
  const bf16* p = base + (size_t)(l & 15) * ld + ((l >> 4) << 3);
  v8bf lo = *(const v8bf*)p;
  v8bf hi = *(const v8bf*)(p + 16);
  v16bf r;
#pragma unroll
  for (int i = 0; i < 8; ++i) { r[i] = lo[i]; r[i + 8] = hi[i]; }
  return r;
}

__device__ inline v8f wmma_bf16(v16bf a, v16bf b, v8f c) {
  return __builtin_amdgcn_wmma_f32_16x16x32_bf16(false, a, false, b,
                                                 (short)0, c, false, false);
}

// Async-copy one 64(N) x 32(K) bf16 weight tile into LDS (padded pitch).
// 256 threads x 16B = 4KB. CDNA5 GLOBAL_LOAD_ASYNC_TO_LDS_B128, GVS mode,
// tracked by ASYNCcnt.
__device__ inline void stage_wtile(bf16* tile, const bf16* W, int n0, int kt) {
  const int t = threadIdx.x;
  const int rowL = t >> 2;          // 0..63
  const int kk = (t & 3) * 8;       // 0,8,16,24
  unsigned voff = (unsigned)((((unsigned)(n0 + rowL)) * H_SIZE + kt * 32 + kk) * 2);
  unsigned lds = (unsigned)(uintptr_t)(tile + rowL * WPITCH + kk);
  unsigned long long sbase = (unsigned long long)(uintptr_t)W;
  asm volatile("global_load_async_to_lds_b128 %0, %1, %2"
               :: "v"(lds), "v"(voff), "s"(sbase) : "memory");
}

// ---------------------------------------------------------------------------
// f32 -> bf16 conversion (x4 vectorized)
// ---------------------------------------------------------------------------
__global__ void cvt_kernel(const float* __restrict__ in, bf16* __restrict__ out,
                           int n4) {
  int i = blockIdx.x * blockDim.x + threadIdx.x;
  if (i < n4) {
    float4 f = ((const float4*)in)[i];
    v4bf o;
    o[0] = (bf16)f.x; o[1] = (bf16)f.y; o[2] = (bf16)f.z; o[3] = (bf16)f.w;
    ((v4bf*)out)[i] = o;
  }
}

// ---------------------------------------------------------------------------
// Projection GEMM: out = A(bf16 [4096][1024]) @ W^T(bf16 [1024][1024]) + bias
// Block: 8 waves, 256(M) x 64(N). Wave: 32(M) x 64(N) = 8 WMMAs per k-step.
// W tile (shared by all waves) double-buffered in LDS via async copies.
// mode 0: bf16 row-major out.  mode 1 (V): transposed per head vt[(b*16+h)*64+d][l].
// ---------------------------------------------------------------------------
__global__ void __launch_bounds__(256)
proj_kernel(const bf16* __restrict__ A, const bf16* __restrict__ W,
            const float* __restrict__ bias, bf16* __restrict__ out, int mode) {
  __shared__ __align__(64) bf16 wtile[2][64 * WPITCH];

  const int wave = threadIdx.x >> 5;
  const int lane = threadIdx.x & 31;
  const int m0 = blockIdx.x * 256 + wave * 32;
  const int n0 = blockIdx.y * 64;

  v8f acc[2][4];
#pragma unroll
  for (int mt = 0; mt < 2; ++mt)
#pragma unroll
    for (int j = 0; j < 4; ++j)
#pragma unroll
      for (int r = 0; r < 8; ++r) acc[mt][j][r] = 0.f;

  const bf16* abase = A + (size_t)m0 * H_SIZE;

  // prologue: prefetch k-step 0
  v16bf a_nxt0 = load_a_frag(abase, H_SIZE);
  v16bf a_nxt1 = load_a_frag(abase + 16 * H_SIZE, H_SIZE);
  stage_wtile(wtile[0], W, n0, 0);

  for (int kt = 0; kt < 32; ++kt) {
    v16bf a0 = a_nxt0, a1 = a_nxt1;
    if (kt + 1 < 32) {
      a_nxt0 = load_a_frag(abase + (kt + 1) * 32, H_SIZE);
      a_nxt1 = load_a_frag(abase + 16 * H_SIZE + (kt + 1) * 32, H_SIZE);
      stage_wtile(wtile[(kt + 1) & 1], W, n0, kt + 1);
      asm volatile("s_wait_asynccnt 0x1" ::: "memory");   // current tile landed
    } else {
      asm volatile("s_wait_asynccnt 0x0" ::: "memory");
    }
    __syncthreads();  // all waves' portions of current tile visible

    const bf16* wt = &wtile[kt & 1][0];
#pragma unroll
    for (int j = 0; j < 4; ++j) {
      v16bf b = load_b_frag(wt + j * 16 * WPITCH, WPITCH);
      acc[0][j] = wmma_bf16(a0, b, acc[0][j]);
      acc[1][j] = wmma_bf16(a1, b, acc[1][j]);
    }
    __syncthreads();  // done reading before this buffer is restaged
  }

  const int rowb = (lane >> 4) << 3;
  const int col = lane & 15;
#pragma unroll
  for (int mt = 0; mt < 2; ++mt) {
#pragma unroll
    for (int j = 0; j < 4; ++j) {
      const int n = n0 + j * 16 + col;
      const float bv = bias[n];
#pragma unroll
      for (int r = 0; r < 8; ++r) {
        const int m = m0 + mt * 16 + rowb + r;
        const float v = acc[mt][j][r] + bv;
        if (mode == 0) {
          out[(size_t)m * H_SIZE + n] = (bf16)v;
        } else {
          const int b = m >> 11, l = m & (LSEQ - 1);
          const int h = n >> 6, d = n & (DK - 1);
          out[(((size_t)(b * NHEADS + h)) * DK + d) * LSEQ + l] = (bf16)v;
        }
      }
    }
  }
}

// ---------------------------------------------------------------------------
// Attention: one block per (query-tile-of-16, head, batch). 8 waves.
// LDS: f32 scores [16][2048] (128KB) + bf16 probs [16][2048] (64KB) + red 4KB.
// ---------------------------------------------------------------------------
__global__ void __launch_bounds__(256)
attn_kernel(const bf16* __restrict__ Qb, const bf16* __restrict__ Kb,
            const bf16* __restrict__ Vt, bf16* __restrict__ Ao) {
  extern __shared__ char smem[];
  float* sc = (float*)smem;                                     // [16][2048]
  bf16* pr = (bf16*)(smem + 16 * LSEQ * 4);                     // [16][2048]
  float* red = (float*)(smem + 16 * LSEQ * 4 + 16 * LSEQ * 2);  // [4][16][16]

  const int wave = threadIdx.x >> 5;
  const int lane = threadIdx.x & 31;
  const int q0 = blockIdx.x * 16;
  const int h = blockIdx.y;
  const int b = blockIdx.z;
  const size_t bh = (size_t)b * NHEADS + h;
  const int rowb = (lane >> 4) << 3;
  const int col = lane & 15;
  const float scale = 0.125f;  // 1/sqrt(64)

  // ---- phase 1: scores = Q K^T * scale (B-fragments register-pipelined) ----
  const bf16* qbase = Qb + ((size_t)(b * LSEQ + q0)) * H_SIZE + h * DK;
  v16bf aq0 = load_a_frag(qbase, H_SIZE);
  v16bf aq1 = load_a_frag(qbase + 32, H_SIZE);

  const bf16* kbase0 = Kb + ((size_t)(b * LSEQ + wave * 256)) * H_SIZE + h * DK;
  v16bf kb_n0 = load_b_frag(kbase0, H_SIZE);
  v16bf kb_n1 = load_b_frag(kbase0 + 32, H_SIZE);

  for (int i = 0; i < 16; ++i) {
    const int nt = wave * 16 + i;  // 128 key tiles over 8 waves
    v16bf k0 = kb_n0, k1 = kb_n1;
    if (i + 1 < 16) {
      const bf16* kb_next =
          Kb + ((size_t)(b * LSEQ + (nt + 1) * 16)) * H_SIZE + h * DK;
      kb_n0 = load_b_frag(kb_next, H_SIZE);
      kb_n1 = load_b_frag(kb_next + 32, H_SIZE);
    }
    v8f acc;
#pragma unroll
    for (int r = 0; r < 8; ++r) acc[r] = 0.f;
    acc = wmma_bf16(aq0, k0, acc);
    acc = wmma_bf16(aq1, k1, acc);
#pragma unroll
    for (int r = 0; r < 8; ++r)
      sc[(size_t)(rowb + r) * LSEQ + nt * 16 + col] = acc[r] * scale;
  }
  __syncthreads();

  // ---- phase 2: exact softmax per row (2 rows per wave, wave32 shuffles) ----
  for (int rr = 0; rr < 2; ++rr) {
    const int row = wave * 2 + rr;
    float* srow = sc + (size_t)row * LSEQ;
    float m = -1e30f;
    for (int j = lane; j < LSEQ; j += 32) m = fmaxf(m, srow[j]);
#pragma unroll
    for (int off = 16; off > 0; off >>= 1) m = fmaxf(m, __shfl_xor(m, off, 32));
    float s = 0.f;
    for (int j = lane; j < LSEQ; j += 32) {
      float e = __expf(srow[j] - m);
      srow[j] = e;
      s += e;
    }
#pragma unroll
    for (int off = 16; off > 0; off >>= 1) s += __shfl_xor(s, off, 32);
    const float inv = 1.0f / s;
    bf16* prow = pr + (size_t)row * LSEQ;
    for (int j = lane; j < LSEQ; j += 32) prow[j] = (bf16)(srow[j] * inv);
  }
  __syncthreads();

  // ---- phase 3: out = P @ V (M=16, N=64 as 4 tiles, K=2048 split 2 ways) ----
  const int j = wave & 3;       // N tile
  const int khalf = wave >> 2;  // K split
  v8f acc;
#pragma unroll
  for (int r = 0; r < 8; ++r) acc[r] = 0.f;
  const bf16* vbase = Vt + (bh * DK + j * 16) * LSEQ;
  const int k0i = khalf * 32, k1i = khalf * 32 + 32;
  v16bf vb_n = load_b_frag(vbase + k0i * 32, LSEQ);
  for (int kt = k0i; kt < k1i; ++kt) {
    v16bf bc = vb_n;
    if (kt + 1 < k1i) vb_n = load_b_frag(vbase + (kt + 1) * 32, LSEQ);
    v16bf a = load_a_frag(pr + kt * 32, LSEQ);
    acc = wmma_bf16(a, bc, acc);
  }
  if (khalf == 1) {
#pragma unroll
    for (int r = 0; r < 8; ++r) red[j * 256 + (rowb + r) * 16 + col] = acc[r];
  }
  __syncthreads();
  if (khalf == 0) {
#pragma unroll
    for (int r = 0; r < 8; ++r) {
      const float v = acc[r] + red[j * 256 + (rowb + r) * 16 + col];
      const int m = q0 + rowb + r;
      Ao[((size_t)(b * LSEQ + m)) * H_SIZE + h * DK + j * 16 + col] = (bf16)v;
    }
  }
}

// ---------------------------------------------------------------------------
// Output projection + residual: y = Ao @ Wo^T + bo + x (f32 out)
// Same pipelined/LDS-staged structure as proj_kernel.
// ---------------------------------------------------------------------------
__global__ void __launch_bounds__(256)
oproj_kernel(const bf16* __restrict__ A, const bf16* __restrict__ W,
             const float* __restrict__ bias, const float* __restrict__ resid,
             float* __restrict__ y) {
  __shared__ __align__(64) bf16 wtile[2][64 * WPITCH];

  const int wave = threadIdx.x >> 5;
  const int lane = threadIdx.x & 31;
  const int m0 = blockIdx.x * 256 + wave * 32;
  const int n0 = blockIdx.y * 64;

  v8f acc[2][4];
#pragma unroll
  for (int mt = 0; mt < 2; ++mt)
#pragma unroll
    for (int j = 0; j < 4; ++j)
#pragma unroll
      for (int r = 0; r < 8; ++r) acc[mt][j][r] = 0.f;

  const bf16* abase = A + (size_t)m0 * H_SIZE;
  v16bf a_nxt0 = load_a_frag(abase, H_SIZE);
  v16bf a_nxt1 = load_a_frag(abase + 16 * H_SIZE, H_SIZE);
  stage_wtile(wtile[0], W, n0, 0);

  for (int kt = 0; kt < 32; ++kt) {
    v16bf a0 = a_nxt0, a1 = a_nxt1;
    if (kt + 1 < 32) {
      a_nxt0 = load_a_frag(abase + (kt + 1) * 32, H_SIZE);
      a_nxt1 = load_a_frag(abase + 16 * H_SIZE + (kt + 1) * 32, H_SIZE);
      stage_wtile(wtile[(kt + 1) & 1], W, n0, kt + 1);
      asm volatile("s_wait_asynccnt 0x1" ::: "memory");
    } else {
      asm volatile("s_wait_asynccnt 0x0" ::: "memory");
    }
    __syncthreads();

    const bf16* wt = &wtile[kt & 1][0];
#pragma unroll
    for (int j = 0; j < 4; ++j) {
      v16bf b = load_b_frag(wt + j * 16 * WPITCH, WPITCH);
      acc[0][j] = wmma_bf16(a0, b, acc[0][j]);
      acc[1][j] = wmma_bf16(a1, b, acc[1][j]);
    }
    __syncthreads();
  }

  const int rowb = (lane >> 4) << 3;
  const int col = lane & 15;
#pragma unroll
  for (int mt = 0; mt < 2; ++mt) {
#pragma unroll
    for (int j = 0; j < 4; ++j) {
      const int n = n0 + j * 16 + col;
      const float bv = bias[n];
#pragma unroll
      for (int r = 0; r < 8; ++r) {
        const size_t idx = (size_t)(m0 + mt * 16 + rowb + r) * H_SIZE + n;
        y[idx] = acc[mt][j][r] + bv + resid[idx];
      }
    }
  }
}

// ---------------------------------------------------------------------------
// Row LayerNorm: out = (y - mu) / sqrt(var + eps) * gamma + beta
// ---------------------------------------------------------------------------
__global__ void __launch_bounds__(256)
ln_kernel(const float* __restrict__ y, const float* __restrict__ gamma,
          const float* __restrict__ beta, float* __restrict__ out) {
  __shared__ float ps[8], ps2[8];
  const int row = blockIdx.x;
  const float* yr = y + (size_t)row * H_SIZE;
  float s = 0.f, s2 = 0.f;
  for (int j = threadIdx.x; j < H_SIZE; j += blockDim.x) {
    const float v = yr[j];
    s += v;
    s2 += v * v;
  }
#pragma unroll
  for (int off = 16; off > 0; off >>= 1) {
    s += __shfl_xor(s, off, 32);
    s2 += __shfl_xor(s2, off, 32);
  }
  const int wave = threadIdx.x >> 5, lane = threadIdx.x & 31;
  if (lane == 0) { ps[wave] = s; ps2[wave] = s2; }
  __syncthreads();
  float ts = 0.f, ts2 = 0.f;
#pragma unroll
  for (int i = 0; i < 8; ++i) { ts += ps[i]; ts2 += ps2[i]; }
  const float mu = ts * (1.0f / H_SIZE);
  const float var = ts2 * (1.0f / H_SIZE) - mu * mu;
  const float inv = rsqrtf(var + 1e-5f);
  for (int j = threadIdx.x; j < H_SIZE; j += blockDim.x)
    out[(size_t)row * H_SIZE + j] = (yr[j] - mu) * inv * gamma[j] + beta[j];
}

// ---------------------------------------------------------------------------
// Host launch
// ---------------------------------------------------------------------------
extern "C" void kernel_launch(void* const* d_in, const int* in_sizes, int n_in,
                              void* d_out, int out_size, void* d_ws,
                              size_t ws_size, hipStream_t stream) {
  const float* x  = (const float*)d_in[0];
  const float* Wq = (const float*)d_in[1];
  const float* bq = (const float*)d_in[2];
  const float* Wk = (const float*)d_in[3];
  const float* bk = (const float*)d_in[4];
  const float* Wv = (const float*)d_in[5];
  const float* bv = (const float*)d_in[6];
  const float* Wo = (const float*)d_in[7];
  const float* bo = (const float*)d_in[8];
  const float* g  = (const float*)d_in[9];
  const float* be = (const float*)d_in[10];
  float* out = (float*)d_out;

  // workspace carving (bytes)
  char* ws = (char*)d_ws;
  bf16* xb  = (bf16*)(ws + 0);            //  8 MB  x bf16 [4096][1024]
  bf16* wqb = (bf16*)(ws + (8u << 20));   //  2 MB
  bf16* wkb = (bf16*)(ws + (10u << 20));  //  2 MB
  bf16* wvb = (bf16*)(ws + (12u << 20));  //  2 MB
  bf16* wob = (bf16*)(ws + (14u << 20));  //  2 MB
  bf16* qb  = (bf16*)(ws + (16u << 20));  //  8 MB  Q bf16 [4096][1024]
  bf16* kb  = (bf16*)(ws + (24u << 20));  //  8 MB  K bf16 [4096][1024]
  bf16* vt  = (bf16*)(ws + (32u << 20));  //  8 MB  V^T bf16 [2*16*64][2048]
  bf16* ao  = (bf16*)(ws + (40u << 20));  //  8 MB  attn out bf16
  float* yf = (float*)(ws + (48u << 20)); // 16 MB  pre-LN f32

  // 1. convert to bf16
  const int NX = BL * H_SIZE / 4;
  const int NW = H_SIZE * H_SIZE / 4;
  cvt_kernel<<<(NX + 255) / 256, 256, 0, stream>>>(x, xb, NX);
  cvt_kernel<<<(NW + 255) / 256, 256, 0, stream>>>(Wq, wqb, NW);
  cvt_kernel<<<(NW + 255) / 256, 256, 0, stream>>>(Wk, wkb, NW);
  cvt_kernel<<<(NW + 255) / 256, 256, 0, stream>>>(Wv, wvb, NW);
  cvt_kernel<<<(NW + 255) / 256, 256, 0, stream>>>(Wo, wob, NW);

  // 2. Q/K/V projections (V stored transposed per head)
  dim3 gproj(BL / 256, H_SIZE / 64);  // (16, 16)
  proj_kernel<<<gproj, 256, 0, stream>>>(xb, wqb, bq, qb, 0);
  proj_kernel<<<gproj, 256, 0, stream>>>(xb, wkb, bk, kb, 0);
  proj_kernel<<<gproj, 256, 0, stream>>>(xb, wvb, bv, vt, 1);

  // 3. attention (exact softmax, scores resident in 320KB CDNA5 LDS)
  const size_t shmem = 16 * LSEQ * 4 + 16 * LSEQ * 2 + 4 * 16 * 16 * 4;  // 200704
  hipFuncSetAttribute((const void*)attn_kernel,
                      hipFuncAttributeMaxDynamicSharedMemorySize, (int)shmem);
  dim3 gattn(LSEQ / 16, NHEADS, BATCH);  // (128, 16, 2)
  attn_kernel<<<gattn, 256, shmem, stream>>>(qb, kb, vt, ao);

  // 4. output projection + residual
  oproj_kernel<<<gproj, 256, 0, stream>>>(ao, wob, bo, x, yf);

  // 5. LayerNorm
  ln_kernel<<<BL, 256, 0, stream>>>(yf, g, be, out);
}
